// Cluster_33354716021218
// MI455X (gfx1250) — matrix operand order repaired and checked by reading, not compiled, not run on hardware
//
#include <hip/hip_runtime.h>
#include <stdint.h>

// ---------------------------------------------------------------------------
// Instance clustering (SpatialEmbeddings-style) for 2048x2048, gfx1250.
// Pipeline:
//   k_zero    : zero counts / dist_map / inst / seed counter
//   k_prep    : seed_map = sigmoid(pred4); scatter-count degridded embeddings
//   k_box     : 3x3 box-sum of counts per 16x16 tile via two
//               v_wmma_f32_16x16x32_f16 (banded matrix), TDM tile staging,
//               threshold -> compact seed list
//   k_cluster : persistent single-workgroup greedy loop (seed list in LDS)
// ---------------------------------------------------------------------------

#define HH 2048
#define WW 2048
#define PLANE (HH * WW)
#define LIST_CAP 65536
#define SEED_CAP 4096
#define WINH 192

typedef _Float16 v16h __attribute__((ext_vector_type(16)));
typedef float    v8f  __attribute__((ext_vector_type(8)));
typedef unsigned int u32x4 __attribute__((ext_vector_type(4)));
typedef int      i32x4 __attribute__((ext_vector_type(4)));
typedef int      i32x8 __attribute__((ext_vector_type(8)));

__device__ __forceinline__ float sigmoidf_(float x) { return 1.0f / (1.0f + expf(-x)); }
__device__ __forceinline__ int reflw(int i) {
    i = (i < 0) ? -i : i;
    return (i > WW - 1) ? (2 * (WW - 1) - i) : i;
}

// ---------------------------------------------------------------------------
__global__ void k_zero(float* counts, float* distm, int* inst, int* scnt) {
    int i = blockIdx.x * blockDim.x + threadIdx.x;
    int st = gridDim.x * blockDim.x;
    for (; i < PLANE; i += st) { counts[i] = 0.0f; distm[i] = 0.0f; inst[i] = 0; }
    if (blockIdx.x == 0 && threadIdx.x == 0) *scnt = 0;
}

// ---------------------------------------------------------------------------
__global__ void k_prep(const float* __restrict__ pred, float* __restrict__ counts,
                       float* __restrict__ seedm) {
    int i = blockIdx.x * blockDim.x + threadIdx.x;
    int st = gridDim.x * blockDim.x;
    const float inv = 1.0f / (float)(WW - 1);
    for (; i < PLANE; i += st) {
        int y = i >> 11, x = i & (WW - 1);
        float sd = sigmoidf_(pred[4 * PLANE + i]);
        seedm[i] = sd;
        if (sd > 0.5f) {  // mask; invalid lanes add 0 in the reference -> skip
            float ey = tanhf(pred[i]) + (float)y * inv;
            float ex = tanhf(pred[PLANE + i]) + (float)x * inv;
            int py = (int)rintf(ey * (float)(HH - 1));
            int px = (int)rintf(ex * (float)(WW - 1));
            if (py >= 0 && py < HH && px >= 0 && px < WW)
                atomicAdd(counts + py * WW + px, 1.0f);  // exact: integer sums
        }
    }
}

// ---------------------------------------------------------------------------
// 3x3 box-sum of `counts` per 16x16 tile: Y = R * X * R  (R = tridiag ones).
// Interior handled with two f16 WMMAs (exact for small-int counts, f32 accum);
// halo ring contributions added scalar from the 18x18 LDS tile.
// LDS buffers are zero-padded past the live region so that every WMMA fragment
// element can be loaded unconditionally (no per-element exec-mask branches).
__global__ __launch_bounds__(32) void k_box(const float* __restrict__ counts,
                                            const float* __restrict__ seedm,
                                            const float* __restrict__ bias_p,
                                            int* __restrict__ seed_list,
                                            int* __restrict__ seed_cnt) {
    __shared__ float t[33 * 18];    // rows 0..17: 18x18 tile; rows 18..32: zeros
    __shared__ float y1s[16][34];   // cols 0..15: Y1; cols 16..33: zeros

    const int lane = threadIdx.x;
    const int tx = blockIdx.x, ty = blockIdx.y;
    const int gx0 = tx * 16 - 1, gy0 = ty * 16 - 1;
    const bool interior = (tx > 0) && (tx < gridDim.x - 1) && (ty > 0) && (ty < gridDim.y - 1);

    // Zero the padding regions (disjoint from the TDM destination rows 0..17).
    for (int k = lane; k < 15 * 18; k += 32) t[18 * 18 + k] = 0.0f;
    for (int k = lane; k < 16 * 18; k += 32) y1s[k / 18][16 + (k % 18)] = 0.0f;

    if (interior) {
        // Tensor Data Mover: 18x18 f32 tile from the counts plane into LDS.
        unsigned long long gaddr =
            (unsigned long long)(uintptr_t)(counts + (long)gy0 * WW + gx0);
        unsigned ldso = (unsigned)(uintptr_t)(void*)t;  // LDS byte offset (addr[31:0])
        u32x4 g0;
        g0[0] = 1u;                                   // count=1 valid descriptor
        g0[1] = ldso;                                 // lds_addr
        g0[2] = (unsigned)gaddr;                      // global_addr[31:0]
        g0[3] = ((unsigned)(gaddr >> 32) & 0x1FFFFFFu) | (2u << 30);  // addr hi | type=2
        i32x8 g1;
        g1[0] = (int)(2u << 16);                                  // data_size = 4B
        g1[1] = (int)((unsigned)WW << 16);                        // tensor_dim0 lo16
        g1[2] = (int)(((unsigned)WW >> 16) | ((unsigned)HH << 16)); // dim0 hi | dim1 lo
        g1[3] = (int)(((unsigned)HH >> 16) | (18u << 16));        // dim1 hi | tile_dim0=18
        g1[4] = (int)18u;                                         // tile_dim1=18, tile_dim2=0
        g1[5] = (int)WW;                                          // tensor_dim0_stride lo32
        g1[6] = 0; g1[7] = 0;
        i32x4 z4 = {0, 0, 0, 0};
        i32x8 z8 = {0, 0, 0, 0, 0, 0, 0, 0};
        __builtin_amdgcn_tensor_load_to_lds(g0, g1, z4, z4, z8, 0);
        __builtin_amdgcn_s_wait_tensorcnt(0);
    } else {
        // Border tiles: scalar loads with reflect indexing (matches jnp reflect pad).
        for (int k = lane; k < 18 * 18; k += 32) {
            int r = k / 18, c = k % 18;
            t[k] = counts[reflw(gy0 + r) * WW + reflw(gx0 + c)];
        }
    }
    __syncthreads();

    const int hl = lane >> 4;      // half-wave select
    const int col = lane & 15;

    // A = R (16x16 tridiagonal band, zero-padded to 16x32). 16-bit A layout:
    // lanes0-15: e<8 -> K=e, e>=8 -> K=e+8 ; lanes16-31: e<8 -> K=e+8, e>=8 -> K=e+16.
    v16h aR;
#pragma unroll
    for (int e = 0; e < 16; ++e) {
        int K = e + 8 * hl + ((e >= 8) ? 8 : 0);
        int d = K - col;
        aR[e] = (_Float16)((K < 16 && d <= 1 && d >= -1) ? 1.0f : 0.0f);
    }
    // B = X (inner 16x16 tile, padded to 32x16): row 16 is the halo row, rows >= 17 read the
    // zeroed LDS pad rows, so loads are unconditional / branch-free.
    v16h bX;
#pragma unroll
    for (int e = 0; e < 16; ++e) {
        int K = e + 8 * hl + ((e >= 8) ? 8 : 0);     // 0..31
        bX[e] = (_Float16)t[(1 + K) * 18 + (1 + col)];
    }
    v8f zero = {};
    v8f c1 = __builtin_amdgcn_wmma_f32_16x16x32_f16(false, aR, false, bX,
                                                    (short)0, zero, false, false);
    // Transpose accumulator frag into an A frag via LDS.
#pragma unroll
    for (int r = 0; r < 8; ++r) y1s[r + 8 * hl][col] = c1[r];
    __syncthreads();

    v16h a2;
#pragma unroll
    for (int e = 0; e < 16; ++e) {
        int K = e + 8 * hl + ((e >= 8) ? 8 : 0);     // 0..31; cols >=16 are zero pad
        a2[e] = (_Float16)y1s[col][K];
    }
    v16h bR;
#pragma unroll
    for (int e = 0; e < 16; ++e) {
        int K = e + 8 * hl + ((e >= 8) ? 8 : 0);
        int d = K - col;
        bR[e] = (_Float16)((K < 16 && d <= 1 && d >= -1) ? 1.0f : 0.0f);
    }
    v8f c2 = __builtin_amdgcn_wmma_f32_16x16x32_f16(false, a2, false, bR,
                                                    (short)0, zero, false, false);

    const float bias9 = 9.0f * (*bias_p);
#pragma unroll
    for (int r = 0; r < 8; ++r) {
        int M = r + 8 * hl, N = col;
        float s = c2[r];
        // Halo-ring corrections (the WMMA saw zero padding at tile edges).
        if (M == 0)  s += t[0 * 18 + N] + t[0 * 18 + N + 1] + t[0 * 18 + N + 2];
        if (M == 15) s += t[17 * 18 + N] + t[17 * 18 + N + 1] + t[17 * 18 + N + 2];
        if (N == 0) {
            if (M != 0)  s += t[M * 18 + 0];
            s += t[(M + 1) * 18 + 0];
            if (M != 15) s += t[(M + 2) * 18 + 0];
        }
        if (N == 15) {
            if (M != 0)  s += t[M * 18 + 17];
            s += t[(M + 1) * 18 + 17];
            if (M != 15) s += t[(M + 2) * 18 + 17];
        }
        int gy = ty * 16 + M, gx = tx * 16 + N, g = gy * WW + gx;
        // smoothed = (box_mean + bias) * 9 = box_sum + 9*bias ; seed needs seed_map>0.1
        if ((s + bias9 >= 4.5f) && (seedm[g] > 0.1f)) {
            int pos = atomicAdd(seed_cnt, 1);
            if (pos < LIST_CAP) seed_list[pos] = g;
        }
    }
}

// ---------------------------------------------------------------------------
// Persistent single-workgroup greedy clustering. Seed list in LDS; each
// iteration evaluates dist only in a +/-WINH window (proposal needs dist>0.5).
__global__ __launch_bounds__(1024) void k_cluster(const float* __restrict__ pred,
                                                  const float* __restrict__ bias_p,
                                                  const int* __restrict__ minsz_p,
                                                  const float* __restrict__ seedm,
                                                  float* __restrict__ distm,
                                                  int* __restrict__ inst,
                                                  const int* __restrict__ seed_list,
                                                  const int* __restrict__ seed_cnt) {
    __shared__ int   sidx[SEED_CAP];
    __shared__ float sval[SEED_CAP];
    __shared__ unsigned char alive[SEED_CAP];
    __shared__ float rv[1024];
    __shared__ int   ri[1024];
    __shared__ int   r1[1024];
    __shared__ int   r2[1024];
    __shared__ float ctr[4];  // cy, cx, s0, s1
    const int tid = threadIdx.x;
    const float inv = 1.0f / (float)(WW - 1);

    int n = *seed_cnt;
    if (n > SEED_CAP) n = SEED_CAP;
    for (int i = tid; i < n; i += 1024) {
        int g = seed_list[i];
        sidx[i] = g; sval[i] = seedm[g]; alive[i] = 1;
    }
    const float bias = *bias_p;
    const int minsz = *minsz_p;
    __syncthreads();

    int count = 1;
    for (int iter = 0; iter < n; ++iter) {
        // --- argmax over alive seeds (tie-break: smallest pixel index) ---
        float bv = -1.0f; int bi = -1; int bg = 0x7fffffff;
        for (int i = tid; i < n; i += 1024) {
            if (alive[i]) {
                float v = sval[i]; int g = sidx[i];
                if (v > bv || (v == bv && g < bg)) { bv = v; bi = i; bg = g; }
            }
        }
        rv[tid] = bv; ri[tid] = bi; r1[tid] = bg;
        __syncthreads();
        for (int off = 512; off > 0; off >>= 1) {
            if (tid < off) {
                float v2 = rv[tid + off];
                if (v2 > rv[tid] || (v2 == rv[tid] && r1[tid + off] < r1[tid])) {
                    rv[tid] = v2; ri[tid] = ri[tid + off]; r1[tid] = r1[tid + off];
                }
            }
            __syncthreads();
        }
        if (rv[0] < 0.0f) break;  // no unclustered seeds remain
        const int best = ri[0];
        const int g = r1[0];
        const int iy = g >> 11, ix = g & (WW - 1);

        if (tid == 0) {
            alive[best] = 0;  // unclustered[iy,ix] = False
            ctr[0] = tanhf(pred[g]) + (float)iy * inv;
            ctr[1] = tanhf(pred[PLANE + g]) + (float)ix * inv;
            for (int c = 0; c < 2; ++c) {  // 3x3 reflect box-mean of sigma + bias
                float acc = 0.0f;
                for (int dy = -1; dy <= 1; ++dy)
                    for (int dx = -1; dx <= 1; ++dx)
                        acc += expf(10.0f * sigmoidf_(
                            pred[(2 + c) * PLANE + reflw(iy + dy) * WW + reflw(ix + dx)]));
                ctr[2 + c] = acc * (1.0f / 9.0f) + bias;
            }
        }
        __syncthreads();
        const float cy = ctr[0], cx = ctr[1], s0 = ctr[2], s1 = ctr[3];
        const int y0 = (iy - WINH > 0) ? iy - WINH : 0;
        const int y1e = (iy + WINH + 1 < HH) ? iy + WINH + 1 : HH;
        const int x0 = (ix - WINH > 0) ? ix - WINH : 0;
        const int x1e = (ix + WINH + 1 < WW) ? ix + WINH + 1 : WW;
        const int wx = x1e - x0, tot = (y1e - y0) * wx;

        // --- pass 1: psum / colliding ---
        int psl = 0, col_l = 0;
        for (int k = tid; k < tot; k += 1024) {
            int y = y0 + k / wx, x = x0 + k % wx, p = y * WW + x;
            float ey = tanhf(pred[p]) + (float)y * inv;
            float ex = tanhf(pred[PLANE + p]) + (float)x * inv;
            float dyv = ey - cy, dxv = ex - cx;
            float dist = expf(-(dyv * dyv * s0 + dxv * dxv * s1));
            bool gt = dist > 0.5f;
            bool prop = gt && (dist > distm[p]) && (seedm[p] > 0.5f);
            psl += prop ? 1 : 0;
            col_l += (gt && inst[p] > 0) ? 1 : 0;
        }
        r1[tid] = psl; r2[tid] = col_l;
        __syncthreads();
        for (int off = 512; off > 0; off >>= 1) {
            if (tid < off) { r1[tid] += r1[tid + off]; r2[tid] += r2[tid + off]; }
            __syncthreads();
        }
        const int psum = r1[0], coll = r2[0];
        const bool accept = (2 * coll < psum) && (psum >= minsz);

        // --- clear seeds covered by accepted proposal (uses OLD dist_map) ---
        if (accept) {
            for (int i = tid; i < n; i += 1024) {
                if (alive[i]) {
                    int sg = sidx[i]; int sy = sg >> 11, sx = sg & (WW - 1);
                    if (sy >= y0 && sy < y1e && sx >= x0 && sx < x1e) {
                        float ey = tanhf(pred[sg]) + (float)sy * inv;
                        float ex = tanhf(pred[PLANE + sg]) + (float)sx * inv;
                        float dyv = ey - cy, dxv = ex - cx;
                        float dist = expf(-(dyv * dyv * s0 + dxv * dxv * s1));
                        if (dist > 0.5f && dist > distm[sg] && seedm[sg] > 0.5f)
                            alive[i] = 0;
                    }
                }
            }
        }
        __threadfence_block();
        __syncthreads();  // seed-clear reads of dist_map done before updates below

        // --- pass 2: update inst (if accept) and dist_map (unconditional) ---
        for (int k = tid; k < tot; k += 1024) {
            int y = y0 + k / wx, x = x0 + k % wx, p = y * WW + x;
            float ey = tanhf(pred[p]) + (float)y * inv;
            float ex = tanhf(pred[PLANE + p]) + (float)x * inv;
            float dyv = ey - cy, dxv = ex - cx;
            float dist = expf(-(dyv * dyv * s0 + dxv * dxv * s1));
            if (dist > 0.5f && dist > distm[p] && seedm[p] > 0.5f) {
                if (accept) inst[p] = count;
                distm[p] = dist;  // reference updates dist_map where proposal regardless
            }
        }
        if (accept) count++;
        __threadfence_block();
        __syncthreads();
    }
}

// ---------------------------------------------------------------------------
extern "C" void kernel_launch(void* const* d_in, const int* in_sizes, int n_in,
                              void* d_out, int out_size, void* d_ws, size_t ws_size,
                              hipStream_t stream) {
    const float* pred   = (const float*)d_in[0];
    const float* bias   = (const float*)d_in[1];
    // d_in[2] = n_sigma (== 2, specialized); d_in[3] = min_obj_size
    const int*   minsz  = (const int*)d_in[3];
    int*         inst   = (int*)d_out;

    char* ws = (char*)d_ws;
    float* counts = (float*)(ws);
    float* seedm  = (float*)(ws + (size_t)PLANE * 4);
    float* distm  = (float*)(ws + (size_t)PLANE * 8);
    int*   slist  = (int*)(ws + (size_t)PLANE * 12);
    int*   scnt   = (int*)(ws + (size_t)PLANE * 12 + (size_t)LIST_CAP * 4);

    k_zero<<<2048, 256, 0, stream>>>(counts, distm, inst, scnt);
    k_prep<<<4096, 256, 0, stream>>>(pred, counts, seedm);
    k_box<<<dim3(WW / 16, HH / 16), 32, 0, stream>>>(counts, seedm, bias, slist, scnt);
    k_cluster<<<1, 1024, 0, stream>>>(pred, bias, minsz, seedm, distm, inst, slist, scnt);
}
